// FlexEquivariantProductBasisBlock_1778116461347
// MI455X (gfx1250) — matrix operand order repaired and checked by reading, not compiled, not run on hardware
//
#include <hip/hip_runtime.h>

typedef __attribute__((ext_vector_type(16))) _Float16 v16h;
typedef __attribute__((ext_vector_type(8)))  _Float16 v8h;
typedef __attribute__((ext_vector_type(8)))  float    v8f;
typedef __attribute__((ext_vector_type(4)))  float    v4f;

#define N_NODES 4096
#define CCH     128
#define NPAIR   (N_NODES * CCH)      // 524288
#define NTILES  (NPAIR / 16)         // 32768 column tiles of 16 pairs
#define MT0     6                    // 0e: 81 rows -> 6 tiles of 16
#define MT1     16                   // 1o: 243 rows -> 16 tiles of 16
#define ROWS_PER_PAIR 356            // 96 + 256 rows, padded to kill LDS bank conflicts
#define WAVES   2                    // waves per block in main kernel
#define SCN     1280                 // NSPEC * CCH

// workspace layout (bytes)
#define OFF_A3     0u                // 22 tiles * 2 ksteps * 512 f16 = 45056 B
#define OFF_WFRAG  45056u            // 2 irreps * 32 frags * 512 f16 = 65536 B
#define OFF_UEFF2  110592u           // 36*9*1280 f32 = 1658880 B   [row][i][s][c]
#define OFF_UEFF1  1769472u          // 4*9*1280 f32 = 184320 B     [o*9+i][s][c]
#define OFF_F0     1953792u          // 524288 f16 = 1048576 B
#define OFF_F1     3002368u          // 3 * 524288 f16 = 3145728 B

// ---------------------------------------------------------------------------
// Prep 1: U3 matrices (d*81 x 36, K zero-padded to 64) into WMMA A-fragment-
// major f16 layout: [tile][kstep][lane][16].  A 16x32 f16 layout used here (contiguous k):
// lane = row m (mod 16); lanes<16 cover k {0..15}, lanes>=16
// cover {16..31}; element e=2j+lo -> VGPR j.
__global__ void prep_u3_frags(const float* __restrict__ U3_0e,
                              const float* __restrict__ U3_1o,
                              _Float16* __restrict__ A3) {
  int tid = blockIdx.x * blockDim.x + threadIdx.x;
  if (tid >= (MT0 + MT1) * 2 * 32) return;
  int lane  = tid & 31;
  int kstep = (tid >> 5) & 1;
  int tg    = tid >> 6;                         // global tile 0..21
  const float* src; int rows; int mt;
  if (tg < MT0) { src = U3_0e; rows = 81;  mt = tg; }
  else          { src = U3_1o; rows = 243; mt = tg - MT0; }
  int m  = mt * 16 + (lane & 15);
  int kb = (lane < 16) ? 0 : 16;
  _Float16* dst = A3 + ((size_t)(tg * 2 + kstep) * 32 + lane) * 16;
#pragma unroll
  for (int e = 0; e < 16; ++e) {
    int j = e >> 1, lo = e & 1;
    int k = kstep * 32 + kb + 2 * j + lo;
    float v = 0.f;
    if (m < rows && k < 36) v = src[m * 36 + k];  // col (i,k3) flat == k
    dst[e] = (_Float16)v;
  }
}

// ---------------------------------------------------------------------------
// Prep 2: Wlin (128x128) into WMMA B-fragment-major f16: per (irrep,kstep,nt)
// 512 f16.  B 32x16 layout: lane = col n (mod 16); lanes<16 hold k=kb..kb+15
// with kb=0, lanes>=16 kb=16; element e -> k = kb + e.
__global__ void prep_w_frags(const float* __restrict__ Wlin_0e,
                             const float* __restrict__ Wlin_1o,
                             _Float16* __restrict__ WF) {
  int tid = blockIdx.x * blockDim.x + threadIdx.x;
  if (tid >= 2 * 4 * 8 * 32) return;
  int lane = tid & 31;
  int nt   = (tid >> 5) & 7;
  int ks   = (tid >> 8) & 3;
  int ir   = tid >> 10;
  const float* W = ir ? Wlin_1o : Wlin_0e;
  int n  = nt * 16 + (lane & 15);
  int kb = ((lane < 16) ? 0 : 16) + ks * 32;
  _Float16* dst = WF + ((size_t)ir * 32 + ks * 8 + nt) * 512 + lane * 16;
#pragma unroll
  for (int e = 0; e < 16; ++e)
    dst[e] = (_Float16)W[(kb + e) * CCH + n];    // B[k][n] = Wlin[k*128+n]
}

// ---------------------------------------------------------------------------
// Prep 3: Ueff2[row(36)][i(9)][s(10)][c(128)] = sum_k U2[o,p,i,k] * W2[s,k,c]
// rows 0..8 = 0e (o=0,p), rows 9..35 = 1o (o,p).  One thread per element:
// both the W2 read and the UE2 write are coalesced in c.
__global__ void prep_ueff2(const float* __restrict__ U2_0e, const float* __restrict__ W2_0e,
                           const float* __restrict__ U2_1o, const float* __restrict__ W2_1o,
                           float* __restrict__ UE2) {
  int tid = blockIdx.x * blockDim.x + threadIdx.x;
  if (tid >= 36 * 9 * SCN) return;
  int c   = tid & 127;
  int s   = (tid >> 7) % 10;
  int i   = (tid / SCN) % 9;
  int row = tid / (9 * SCN);
  float v;
  if (row < 9) {
    int p = row;
    v = U2_0e[p * 18 + i * 2 + 0] * W2_0e[(s * 2 + 0) * CCH + c] +
        U2_0e[p * 18 + i * 2 + 1] * W2_0e[(s * 2 + 1) * CCH + c];
  } else {
    int r = row - 9;
    int o = r / 9, p = r % 9;
    v = U2_1o[o * 162 + p * 18 + i * 2 + 0] * W2_1o[(s * 2 + 0) * CCH + c] +
        U2_1o[o * 162 + p * 18 + i * 2 + 1] * W2_1o[(s * 2 + 1) * CCH + c];
  }
  UE2[tid] = v;
}

// Prep 4: Ueff1[o*9+i][s][c] = U1[o,i,0] * W1[s,0,c], coalesced in c.
__global__ void prep_ueff1(const float* __restrict__ U1_0e, const float* __restrict__ W1_0e,
                           const float* __restrict__ U1_1o, const float* __restrict__ W1_1o,
                           float* __restrict__ UE1) {
  int tid = blockIdx.x * blockDim.x + threadIdx.x;
  if (tid >= 4 * 9 * SCN) return;
  int c = tid & 127;
  int s = (tid >> 7) % 10;
  int i = (tid / SCN) % 9;
  int o = tid / (9 * SCN);
  float v = (o == 0) ? U1_0e[i] * W1_0e[s * CCH + c]
                     : U1_1o[(o - 1) * 9 + i] * W1_1o[s * CCH + c];
  UE1[tid] = v;
}

// ---------------------------------------------------------------------------
// Main: one wave handles 16 (b,c) pairs (all sharing node b).
// Phase 1: out1 = U3 @ (x (x) w3) via v_wmma_f32_16x16x32_f16, D tiles -> LDS.
// Phase 2: per-lane VALU finish of steps 2-3, f -> f16 workspace.
__global__ __launch_bounds__(WAVES * 32)
void sym_contract(const float* __restrict__ x, const int* __restrict__ specie,
                  const _Float16* __restrict__ A3,
                  const float* __restrict__ W3a, const float* __restrict__ W3b,
                  const float* __restrict__ UE2, const float* __restrict__ UE1,
                  _Float16* __restrict__ f0, _Float16* __restrict__ f1) {
  __shared__ __align__(16) float lds[WAVES * 16 * ROWS_PER_PAIR];
  int wave = threadIdx.x >> 5, lane = threadIdx.x & 31;
  int t   = blockIdx.x * WAVES + wave;           // pair-tile id
  int b   = t >> 3;                              // 8 tiles per node (C/16)
  int c0  = (t & 7) << 4;
  int col = c0 + (lane & 15);
  int s   = specie[b];

  float x9[9];
  const float* xp = x + ((size_t)b * CCH + col) * 9;
#pragma unroll
  for (int i = 0; i < 9; ++i) x9[i] = xp[i];
  float w3a[4], w3b[4];
#pragma unroll
  for (int k = 0; k < 4; ++k) {
    w3a[k] = W3a[(s * 4 + k) * CCH + col];
    w3b[k] = W3b[(s * 4 + k) * CCH + col];
  }
  int scbase = s * CCH + col;                    // lane-coalesced (stride 1 in c)
  __builtin_prefetch(UE2 + scbase, 0, 0);        // warm L2 for phase 2
  __builtin_prefetch(UE1 + scbase, 0, 0);

  // B fragments: z[(i,k3)] = x_i * w3_k3, K padded 36 -> 64 (2 ksteps)
  v16h Ba0, Ba1, Bb0, Bb1;
  int kb = (lane & 16) ? 16 : 0;
#pragma unroll
  for (int e = 0; e < 16; ++e) {
    int kk = kb + e;
    int i = kk >> 2, k3 = kk & 3;
    Ba0[e] = (_Float16)(x9[i] * w3a[k3]);
    Bb0[e] = (_Float16)(x9[i] * w3b[k3]);
    bool valid = (lane < 16) && (e < 4);         // kstep1: only k=32..35 live
    Ba1[e] = valid ? (_Float16)(x9[8] * w3a[e]) : (_Float16)0.f;
    Bb1[e] = valid ? (_Float16)(x9[8] * w3b[e]) : (_Float16)0.f;
  }

  float* myL = lds + wave * (16 * ROWS_PER_PAIR) + (lane & 15) * ROWS_PER_PAIR;
  int rsub = (lane & 16) ? 8 : 0;                // D-fragment row half

  // 0e: 6 M-tiles
#pragma unroll
  for (int mt = 0; mt < MT0; ++mt) {
    v16h A0 = *(const v16h*)(A3 + (size_t)(mt * 2 + 0) * 512 + lane * 16);
    v16h A1 = *(const v16h*)(A3 + (size_t)(mt * 2 + 1) * 512 + lane * 16);
    v8f acc = {};
    acc = __builtin_amdgcn_wmma_f32_16x16x32_f16(false, A0, false, Ba0, (short)0, acc, false, false);
    acc = __builtin_amdgcn_wmma_f32_16x16x32_f16(false, A1, false, Ba1, (short)0, acc, false, false);
    float* d = myL + mt * 16 + rsub;
    v4f lo = {acc[0], acc[1], acc[2], acc[3]};
    v4f hi = {acc[4], acc[5], acc[6], acc[7]};
    *(v4f*)(d)     = lo;
    *(v4f*)(d + 4) = hi;
  }
  // 1o: 16 M-tiles (rows base 96 in LDS)
#pragma unroll
  for (int mt = 0; mt < MT1; ++mt) {
    v16h A0 = *(const v16h*)(A3 + (size_t)(MT0 * 2 + mt * 2 + 0) * 512 + lane * 16);
    v16h A1 = *(const v16h*)(A3 + (size_t)(MT0 * 2 + mt * 2 + 1) * 512 + lane * 16);
    v8f acc = {};
    acc = __builtin_amdgcn_wmma_f32_16x16x32_f16(false, A0, false, Bb0, (short)0, acc, false, false);
    acc = __builtin_amdgcn_wmma_f32_16x16x32_f16(false, A1, false, Bb1, (short)0, acc, false, false);
    float* d = myL + 96 + mt * 16 + rsub;
    v4f lo = {acc[0], acc[1], acc[2], acc[3]};
    v4f hi = {acc[4], acc[5], acc[6], acc[7]};
    *(v4f*)(d)     = lo;
    *(v4f*)(d + 4) = hi;
  }

  asm volatile("s_wait_dscnt 0" ::: "memory");   // LDS stores visible to our loads

  // Phase 2: lane handles pair (lane&15), o-groups {2h, 2h+1}
  int h = lane >> 4;
#pragma unroll
  for (int g = 0; g < 2; ++g) {
    int og = h * 2 + g;                          // 0:0e  1..3:1o o=og-1
    int ldsBase = (og == 0) ? 0 : (96 + (og - 1) * 81);
    float f = 0.f;
#pragma unroll
    for (int p2 = 0; p2 < 9; ++p2) {
      const float* l1 = myL + ldsBase + p2 * 9;
      int row = og * 9 + p2;
      float o2 = 0.f;
#pragma unroll
      for (int q = 0; q < 9; ++q) {
        float ue = UE2[(size_t)(row * 9 + q) * SCN + scbase];  // coalesced in c
        o2 = fmaf(l1[q] + ue, x9[q], o2);
      }
      float u1 = UE1[(size_t)(og * 9 + p2) * SCN + scbase];    // coalesced in c
      f = fmaf(o2 + u1, x9[p2], f);
    }
    size_t oidx = (size_t)b * CCH + col;
    if (og == 0) f0[oidx] = (_Float16)f;
    else         f1[(size_t)(og - 1) * NPAIR + oidx] = (_Float16)f;
  }
}

// ---------------------------------------------------------------------------
// Linear: 4 GEMMs (0e; 1o o=0,1,2), each (4096x128)@(128x128) in f16 WMMA.
// One wave per 16-row M-tile; B fragments preformatted in workspace.
__global__ __launch_bounds__(64)
void linear_out(const _Float16* __restrict__ f0, const _Float16* __restrict__ f1,
                const _Float16* __restrict__ WF, const float* __restrict__ bias,
                float* __restrict__ out) {
  int wave = threadIdx.x >> 5, lane = threadIdx.x & 31;
  int task  = blockIdx.x * 2 + wave;             // 0..1023
  int g     = task >> 8;                         // 0:0e  1..3:1o o=g-1
  int mtile = task & 255;
  const _Float16* A  = (g == 0) ? f0 : (f1 + (size_t)(g - 1) * NPAIR);
  const _Float16* WB = WF + (size_t)((g == 0) ? 0 : 1) * 32 * 512;
  int m  = mtile * 16 + (lane & 15);
  int kb = (lane < 16) ? 0 : 16;

  v8f acc[8];
#pragma unroll
  for (int nt = 0; nt < 8; ++nt) { v8f z = {}; acc[nt] = z; }

#pragma unroll
  for (int ks = 0; ks < 4; ++ks) {
    const _Float16* ap = A + (size_t)m * CCH + ks * 32 + kb;
    v8h lo = *(const v8h*)(ap);
    v8h hi = *(const v8h*)(ap + 8);
    v16h a;
#pragma unroll
    for (int e = 0; e < 8; ++e) { a[e] = lo[e]; a[8 + e] = hi[e]; }
#pragma unroll
    for (int nt = 0; nt < 8; ++nt) {
      v16h bf = *(const v16h*)(WB + (size_t)(ks * 8 + nt) * 512 + lane * 16);
      acc[nt] = __builtin_amdgcn_wmma_f32_16x16x32_f16(false, a, false, bf, (short)0, acc[nt], false, false);
    }
  }

  const float scale = 0.08838834764831845f;      // 1/sqrt(128)
  int rowbase = mtile * 16 + ((lane & 16) ? 8 : 0);
#pragma unroll
  for (int nt = 0; nt < 8; ++nt) {
    int mcol = nt * 16 + (lane & 15);
    float bv = (g == 0) ? bias[mcol] : 0.f;
#pragma unroll
    for (int r = 0; r < 8; ++r) {
      float v = acc[nt][r] * scale + bv;
      int node = rowbase + r;
      size_t oidx = (g == 0) ? ((size_t)node * 512 + mcol)
                             : ((size_t)node * 512 + 128 + (size_t)mcol * 3 + (g - 1));
      out[oidx] = v;
    }
  }
}

// ---------------------------------------------------------------------------
extern "C" void kernel_launch(void* const* d_in, const int* in_sizes, int n_in,
                              void* d_out, int out_size, void* d_ws, size_t ws_size,
                              hipStream_t stream) {
  const float* x       = (const float*)d_in[0];
  const int*   specie  = (const int*)  d_in[1];
  const float* U3_0e   = (const float*)d_in[2];
  const float* U2_0e   = (const float*)d_in[3];
  const float* U1_0e   = (const float*)d_in[4];
  const float* W3_0e   = (const float*)d_in[5];
  const float* W2_0e   = (const float*)d_in[6];
  const float* W1_0e   = (const float*)d_in[7];
  const float* U3_1o   = (const float*)d_in[8];
  const float* U2_1o   = (const float*)d_in[9];
  const float* U1_1o   = (const float*)d_in[10];
  const float* W3_1o   = (const float*)d_in[11];
  const float* W2_1o   = (const float*)d_in[12];
  const float* W1_1o   = (const float*)d_in[13];
  const float* Wlin_0e = (const float*)d_in[14];
  const float* Wlin_1o = (const float*)d_in[15];
  const float* bias_0e = (const float*)d_in[16];
  float* out = (float*)d_out;

  char* ws = (char*)d_ws;
  _Float16* A3 = (_Float16*)(ws + OFF_A3);
  _Float16* WF = (_Float16*)(ws + OFF_WFRAG);
  float*    UE2 = (float*)(ws + OFF_UEFF2);
  float*    UE1 = (float*)(ws + OFF_UEFF1);
  _Float16* f0 = (_Float16*)(ws + OFF_F0);
  _Float16* f1 = (_Float16*)(ws + OFF_F1);

  prep_u3_frags<<<11, 128, 0, stream>>>(U3_0e, U3_1o, A3);
  prep_w_frags <<<16, 128, 0, stream>>>(Wlin_0e, Wlin_1o, WF);
  prep_ueff2   <<<3240, 128, 0, stream>>>(U2_0e, W2_0e, U2_1o, W2_1o, UE2);
  prep_ueff1   <<<360, 128, 0, stream>>>(U1_0e, W1_0e, U1_1o, W1_1o, UE1);

  sym_contract<<<NTILES / WAVES, WAVES * 32, 0, stream>>>(
      x, specie, A3, W3_0e, W3_1o, UE2, UE1, f0, f1);

  linear_out<<<512, 64, 0, stream>>>(f0, f1, WF, bias_0e, out);
}
